// Retention_81080392614642
// MI455X (gfx1250) — compile-verified, matrix operand order checked by reading, not run on hardware
//
#include <hip/hip_runtime.h>
#include <hip/hip_bf16.h>

// ---------------------------------------------------------------------------
// Retention layer for MI455X (gfx1250, wave32, WMMA + TDM + async-LDS).
//  - bf16 WMMA w/ fp32 accum for all GEMMs
//  - decay mask recomputed analytically (avoids 256MB mask reads)
//  - single-pass chunked attention: (qk/denom)@V == (qk@V)/denom
//  - GEMM tiles staged with global_load_async_to_lds_b128 (double-buffered)
//  - attention Q/K/V^T tiles staged with the Tensor Data Mover
// ---------------------------------------------------------------------------

typedef __attribute__((ext_vector_type(16))) __bf16 v16bf;
typedef __attribute__((ext_vector_type(8)))  float  v8f;
typedef __attribute__((ext_vector_type(4)))  unsigned int u32x4;
typedef __attribute__((ext_vector_type(8)))  int          i32x8;
typedef __attribute__((ext_vector_type(4)))  int          i32x4;

constexpr int Bc = 2, Lc = 2048, Dc = 1024, Hc = 16, VFc = 2;
constexpr int DKc = 64, DVc = 128;
constexpr int Mtok = Bc * Lc;            // 4096 tokens
constexpr float LN_EPS = 1e-5f;

__device__ inline v8f wmma_bf16(v16bf a, v16bf b, v8f c) {
  return __builtin_amdgcn_wmma_f32_16x16x32_bf16(
      false, a, false, b, (short)0, c, false, false);
}

// LDS generic pointer -> 32-bit LDS address (keeps the low 32 bits of the generic address)
__device__ inline unsigned lds_lo32(const void* p) {
  return (unsigned)(uintptr_t)p;
}

// Async global -> LDS copy, 16 bytes per lane. Tracked with ASYNCcnt.
__device__ inline void async_g2l_b128(unsigned lds, unsigned long long base,
                                      unsigned voff) {
  asm volatile("global_load_async_to_lds_b128 %0, %1, %2"
               :: "v"(lds), "v"(voff), "s"(base) : "memory");
}
__device__ inline void wait_async_le4() {
  asm volatile("s_wait_asynccnt 0x4" ::: "memory");
}
__device__ inline void wait_async_0() {
  asm volatile("s_wait_asynccnt 0x0" ::: "memory");
}

// Tensor Data Mover: 2D tile load (bf16 elements) described by a D#.
// tile_d0 contiguous elems per row, tile_d1 rows, row stride = stride0 elems.
__device__ inline void tdm_load_2d_bf16(unsigned lds, const void* gptr,
                                        unsigned tile_d0, unsigned tile_d1,
                                        unsigned tensor_d0, unsigned tensor_d1,
                                        unsigned stride0) {
  unsigned long long ga = (unsigned long long)(uintptr_t)gptr;
  u32x4 g0;
  g0[0] = 1u;                                   // count=1, user mode
  g0[1] = lds;                                  // lds_addr
  g0[2] = (unsigned)ga;                         // global_addr[31:0]
  g0[3] = (unsigned)(ga >> 32) | (2u << 30);    // global_addr[56:32] | type=2
  i32x8 g1;
  g1[0] = (int)(1u << 16);                      // wg_mask=0, data_size=1 (2B)
  g1[1] = (int)((tensor_d0 & 0xFFFFu) << 16);   // tensor_dim0[15:0] @ bits 63:48
  g1[2] = (int)((tensor_d0 >> 16) | ((tensor_d1 & 0xFFFFu) << 16));
  g1[3] = (int)((tensor_d1 >> 16) | (tile_d0 << 16));   // tile_dim0 @ 127:112
  g1[4] = (int)(tile_d1 & 0xFFFFu);             // tile_dim1 @ 143:128, tile2=0
  g1[5] = (int)stride0;                         // tensor_dim0_stride[31:0]
  g1[6] = 0;                                    // stride0 hi, stride1 lo
  g1[7] = 0;
  i32x4 z4 = {0, 0, 0, 0};
#if defined(__clang_major__) && __clang_major__ >= 23
  i32x8 z8 = {0, 0, 0, 0, 0, 0, 0, 0};
  __builtin_amdgcn_tensor_load_to_lds(g0, g1, z4, z4, z8, 0);
#else
  __builtin_amdgcn_tensor_load_to_lds(g0, g1, z4, z4, 0);
#endif
}

// 16x32 bf16 operand fragment from LDS (ISA 7.12.2 layout): lane holds row
// (lane&15); two contiguous 16B chunks at K = (lane>>4)*8 and 16+(lane>>4)*8.
__device__ inline v16bf ld_frag(const __bf16* p0, int stride, int row0, int kBase) {
  const int lane = threadIdx.x & 31;
  const __bf16* p = p0 + (row0 + (lane & 15)) * stride + kBase + ((lane >> 4) << 3);
  v16bf f;
  ((uint4*)&f)[0] = *(const uint4*)(p);
  ((uint4*)&f)[1] = *(const uint4*)(p + 16);
  return f;
}

// ---------------------------------------------------------------------------
__global__ void cvt_f32_bf16(const float* __restrict__ in, __bf16* __restrict__ out, int n) {
  for (int i = blockIdx.x * blockDim.x + threadIdx.x; i < n; i += gridDim.x * blockDim.x)
    out[i] = (__bf16)in[i];
}

// ---------------------------------------------------------------------------
// Tiled WMMA GEMM: C[M,N] = A[M,K] @ B[N,K]^T; bf16 in, fp32 accum.
// 128x128 tile, k-step 32, 8 waves; async-to-LDS staging, double-buffered.
// MODE 0: bf16 row-major (q,k,g)  MODE 1: bf16 (B,H,DV,L) V^T  MODE 2: fp32.
// ---------------------------------------------------------------------------
template <int MODE>
__global__ __launch_bounds__(256) void gemm_bf16_kernel(
    const __bf16* __restrict__ A, const __bf16* __restrict__ Bm,
    const float* __restrict__ bias, void* __restrict__ Cout,
    int M, int N, int K, float scale) {
  __shared__ __align__(16) __bf16 sA[2][128][32];
  __shared__ __align__(16) __bf16 sB[2][128][32];
  const int tid = threadIdx.x, lane = tid & 31, wave = tid >> 5;
  const int mwave = wave >> 2;   // 0..1
  const int nwave = wave & 3;    // 0..3
  const int rowA0 = blockIdx.y * 128;
  const int rowB0 = blockIdx.x * 128;
  const unsigned long long Ab = (unsigned long long)(uintptr_t)A;
  const unsigned long long Bb = (unsigned long long)(uintptr_t)Bm;

  v8f acc[4][2] = {};

  auto stage = [&](int buf, int k0) {
#pragma unroll
    for (int i = 0; i < 2; ++i) {           // 4 async instrs/wave per stage
      int linear = tid + i * 256;
      int r = linear >> 2, seg = (linear & 3) << 3;
      async_g2l_b128(lds_lo32(&sA[buf][r][seg]), Ab,
                     (unsigned)(((size_t)(rowA0 + r) * K + k0 + seg) * 2));
      async_g2l_b128(lds_lo32(&sB[buf][r][seg]), Bb,
                     (unsigned)(((size_t)(rowB0 + r) * K + k0 + seg) * 2));
    }
  };

  const int nk = K / 32;
  stage(0, 0);
  for (int ik = 0; ik < nk; ++ik) {
    const int buf = ik & 1;
    if (ik + 1 < nk) {
      stage(buf ^ 1, (ik + 1) * 32);        // overlap next tile with compute
      wait_async_le4();                     // current tile landed (in-order)
    } else {
      wait_async_0();
    }
    __syncthreads();
    v16bf af[4], bfr[2];
#pragma unroll
    for (int mi = 0; mi < 4; ++mi)
      af[mi] = ld_frag(&sA[buf][0][0], 32, mwave * 64 + mi * 16, 0);
#pragma unroll
    for (int nj = 0; nj < 2; ++nj)
      bfr[nj] = ld_frag(&sB[buf][0][0], 32, nwave * 32 + nj * 16, 0);
#pragma unroll
    for (int mi = 0; mi < 4; ++mi)
#pragma unroll
      for (int nj = 0; nj < 2; ++nj)
        acc[mi][nj] = wmma_bf16(af[mi], bfr[nj], acc[mi][nj]);
    __syncthreads();                        // done reading buf -> reusable
  }

#pragma unroll
  for (int mi = 0; mi < 4; ++mi)
#pragma unroll
    for (int nj = 0; nj < 2; ++nj)
#pragma unroll
      for (int r = 0; r < 8; ++r) {
        int m = rowA0 + mwave * 64 + mi * 16 + r + ((lane >> 4) << 3);
        int n = rowB0 + nwave * 32 + nj * 16 + (lane & 15);
        float v = (acc[mi][nj][r] + bias[n]) * scale;
        if (MODE == 0) {
          ((__bf16*)Cout)[(size_t)m * N + n] = (__bf16)v;
        } else if (MODE == 1) {
          int b = m / Lc, l = m % Lc;
          int h = n / DVc, dvi = n % DVc;
          ((__bf16*)Cout)[((size_t)(b * Hc + h) * DVc + dvi) * Lc + l] = (__bf16)v;
        } else {
          ((float*)Cout)[(size_t)m * N + n] = v;
        }
      }
}

// ---------------------------------------------------------------------------
// Rotary (theta-shift): (B,L,D) bf16 -> (B,H,L,DK) bf16.
// ---------------------------------------------------------------------------
__global__ void rotary_kernel(const __bf16* __restrict__ in, const float* __restrict__ sinp,
                              const float* __restrict__ cosp, __bf16* __restrict__ out) {
  int idx = blockIdx.x * blockDim.x + threadIdx.x;
  if (idx >= Mtok * Dc / 2) return;
  int f2 = idx % (Dc / 2);
  int bl = idx / (Dc / 2);
  int l = bl % Lc, b = bl / Lc;
  int feat = f2 * 2;
  int h = feat / DKc, t = feat % DKc;
  float t0 = (float)in[(size_t)bl * Dc + feat];
  float t1 = (float)in[(size_t)bl * Dc + feat + 1];
  float c0 = cosp[l * DKc + t],     s0 = sinp[l * DKc + t];
  float c1 = cosp[l * DKc + t + 1], s1 = sinp[l * DKc + t + 1];
  size_t ob = ((size_t)(b * Hc + h) * Lc + l) * DKc + t;
  out[ob]     = (__bf16)(t0 * c0 - t1 * s0);
  out[ob + 1] = (__bf16)(t1 * c1 + t0 * s1);
}

// ---------------------------------------------------------------------------
// Chunked retention attention. Block = one (b, h, 64-row query chunk).
// TDM stages Q/K (1D 8KB tiles) and V^T (2D 128x64 tile, stride L).
// ---------------------------------------------------------------------------
__global__ __launch_bounds__(256) void retention_attn_kernel(
    const __bf16* __restrict__ qr, const __bf16* __restrict__ kr,
    const __bf16* __restrict__ vt, const __bf16* __restrict__ g,
    const float* __restrict__ ln_g, const float* __restrict__ ln_b,
    __bf16* __restrict__ act) {
  __shared__ __align__(16) char smem[41216];
  __bf16* sQ  = (__bf16*)(smem);            // 64x64   (8KB)
  __bf16* sK  = (__bf16*)(smem + 8192);     // 64x64   (8KB)
  __bf16* sVT = (__bf16*)(smem + 16384);    // 128x64  (16KB)
  __bf16* sS  = (__bf16*)(smem + 32768);    // 64x64   (8KB)
  float*  sO  = (float*) (smem + 8192);     // 64x128  (32KB, overlap: post-loop)
  float*  sRow = (float*)(smem + 40960);    // 64 fp32

  const int qi = blockIdx.x, h = blockIdx.y, b = blockIdx.z;
  const int tid = threadIdx.x, lane = tid & 31, wave = tid >> 5;
  const size_t headQK = (size_t)(b * Hc + h) * Lc * DKc;
  const size_t headV  = (size_t)(b * Hc + h) * DVc * Lc;

  if (wave == 0) {                          // TDM: Q chunk (1D, 4096 elems)
    tdm_load_2d_bf16(lds_lo32(sQ), qr + headQK + (size_t)qi * 64 * DKc,
                     /*tile*/ 4096, 1, /*tensor*/ 4096, 1, 4096);
  }
  if (tid < 64) sRow[tid] = 0.f;

  const float dec  = 1.f - exp2f(-5.f - (float)h);
  const float logd = log2f(dec);

  const int smi = (wave & 3) * 16;          // S: rows smi, cols snj..+31
  const int snj = (wave >> 2) * 32;
  const int omi = smi;                      // O: rows omi, cols onj..+63
  const int onj = (wave >> 2) * 64;

  v8f oacc[4] = {};

  for (int j = 0; j <= qi; ++j) {
    __syncthreads();                        // prior WMMA reads of sK/sVT/sS done
    if (wave == 0) {                        // TDM: K chunk (1D) + V^T tile (2D)
      tdm_load_2d_bf16(lds_lo32(sK), kr + headQK + (size_t)j * 64 * DKc,
                       4096, 1, 4096, 1, 4096);
      tdm_load_2d_bf16(lds_lo32(sVT), vt + headV + (size_t)j * 64,
                       /*tile*/ 64, 128, /*tensor*/ (unsigned)Lc, 128,
                       (unsigned)Lc);
      __builtin_amdgcn_s_wait_tensorcnt(0); // covers Q on first iteration too
    }
    __syncthreads();                        // publish staged tiles to all waves

    // S = Q K^T : 2 tiles/wave, 2 k-steps
    v8f sacc[2] = {};
#pragma unroll
    for (int ks = 0; ks < 64; ks += 32) {
      v16bf aq = ld_frag(sQ, 64, smi, ks);
#pragma unroll
      for (int nj2 = 0; nj2 < 2; ++nj2) {
        v16bf kb2 = ld_frag(sK, 64, snj + nj2 * 16, ks);
        sacc[nj2] = wmma_bf16(aq, kb2, sacc[nj2]);
      }
    }
    // analytic decay mask + row-sum (ds_add_f32) + spill S to LDS as bf16
#pragma unroll
    for (int nj2 = 0; nj2 < 2; ++nj2)
#pragma unroll
      for (int r = 0; r < 8; ++r) {
        int m = smi + r + ((lane >> 4) << 3);
        int n = snj + nj2 * 16 + (lane & 15);
        int gi = qi * 64 + m, gj = j * 64 + n;
        float s = sacc[nj2][r];
        s = (gi >= gj) ? s * exp2f(logd * (float)(gi - gj)) : 0.f;
        sS[m * 64 + n] = (__bf16)s;
        atomicAdd(&sRow[m], s);
      }
    __syncthreads();

    // O += S V : 4 tiles/wave, 2 k-steps
#pragma unroll
    for (int ks = 0; ks < 64; ks += 32) {
      v16bf as = ld_frag(sS, 64, omi, ks);
#pragma unroll
      for (int t = 0; t < 4; ++t) {
        v16bf vb2 = ld_frag(sVT, 64, onj + t * 16, ks);
        oacc[t] = wmma_bf16(as, vb2, oacc[t]);
      }
    }
  }
  __syncthreads();

  // denom divide, spill O to LDS (overlap region now free)
#pragma unroll
  for (int t = 0; t < 4; ++t)
#pragma unroll
    for (int r = 0; r < 8; ++r) {
      int m = omi + r + ((lane >> 4) << 3);
      int n = onj + t * 16 + (lane & 15);
      float denom = fabsf(sRow[m]);
      denom = denom > 1.f ? denom : 1.f;
      sO[m * DVc + n] = oacc[t][r] / denom;
    }
  __syncthreads();

  // fused LayerNorm(dv=128) + SiLU(g) gating
  if (tid < 64) {
    int m = tid;
    float mu = 0.f;
    for (int n = 0; n < DVc; ++n) mu += sO[m * DVc + n];
    mu *= (1.f / DVc);
    float var = 0.f;
    for (int n = 0; n < DVc; ++n) { float d = sO[m * DVc + n] - mu; var += d * d; }
    var *= (1.f / DVc);
    float inv = rsqrtf(var + LN_EPS);
    size_t row = (size_t)(b * Lc + qi * 64 + m) * (Hc * DVc) + (size_t)h * DVc;
    for (int n = 0; n < DVc; ++n) {
      float y  = (sO[m * DVc + n] - mu) * inv * ln_g[n] + ln_b[n];
      float gv = (float)g[row + n];
      float si = gv / (1.f + expf(-gv));
      act[row + n] = (__bf16)(si * y);
    }
  }
}

// ---------------------------------------------------------------------------
extern "C" void kernel_launch(void* const* d_in, const int* in_sizes, int n_in,
                              void* d_out, int out_size, void* d_ws, size_t ws_size,
                              hipStream_t stream) {
  (void)in_sizes; (void)n_in; (void)out_size; (void)ws_size;
  const float* x    = (const float*)d_in[0];
  const float* sinp = (const float*)d_in[1];
  const float* cosp = (const float*)d_in[2];
  /* d_in[3] = mask: recomputed analytically on the fly */
  const float* Wq = (const float*)d_in[4];  const float* bq = (const float*)d_in[5];
  const float* Wk = (const float*)d_in[6];  const float* bk = (const float*)d_in[7];
  const float* Wv = (const float*)d_in[8];  const float* bv = (const float*)d_in[9];
  const float* Wg = (const float*)d_in[10]; const float* bg = (const float*)d_in[11];
  const float* Wo = (const float*)d_in[12]; const float* bo = (const float*)d_in[13];
  const float* lng = (const float*)d_in[14]; const float* lnb = (const float*)d_in[15];
  float* out = (float*)d_out;

  char* ws = (char*)d_ws;
  size_t off = 0;
  auto take = [&](size_t elems) {
    __bf16* p = (__bf16*)(ws + off);
    off += (elems * sizeof(__bf16) + 255) & ~(size_t)255;
    return p;
  };
  __bf16* xb  = take((size_t)Mtok * Dc);
  __bf16* Wqb = take((size_t)Dc * Dc);
  __bf16* Wkb = take((size_t)Dc * Dc);
  __bf16* Wvb = take((size_t)Dc * VFc * Dc);
  __bf16* Wgb = take((size_t)Dc * VFc * Dc);
  __bf16* Wob = take((size_t)Dc * Dc * VFc);
  __bf16* qb  = take((size_t)Mtok * Dc);
  __bf16* kb  = take((size_t)Mtok * Dc);
  __bf16* gb  = take((size_t)Mtok * Dc * VFc);
  __bf16* qrb = take((size_t)Mtok * Dc);
  __bf16* krb = take((size_t)Mtok * Dc);
  __bf16* vtb = take((size_t)Mtok * Dc * VFc);     // (B,H,DV,L) transposed V
  __bf16* actb= take((size_t)Mtok * Dc * VFc);

  cvt_f32_bf16<<<512, 256, 0, stream>>>(x,  xb,  Mtok * Dc);
  cvt_f32_bf16<<<512, 256, 0, stream>>>(Wq, Wqb, Dc * Dc);
  cvt_f32_bf16<<<512, 256, 0, stream>>>(Wk, Wkb, Dc * Dc);
  cvt_f32_bf16<<<512, 256, 0, stream>>>(Wv, Wvb, Dc * VFc * Dc);
  cvt_f32_bf16<<<512, 256, 0, stream>>>(Wg, Wgb, Dc * VFc * Dc);
  cvt_f32_bf16<<<512, 256, 0, stream>>>(Wo, Wob, Dc * Dc * VFc);

  const float kscale = 0.125f;  // DK^-0.5
  dim3 blk(256);
  gemm_bf16_kernel<0><<<dim3(Dc / 128, Mtok / 128), blk, 0, stream>>>(
      xb, Wqb, bq, qb, Mtok, Dc, Dc, 1.f);
  gemm_bf16_kernel<0><<<dim3(Dc / 128, Mtok / 128), blk, 0, stream>>>(
      xb, Wkb, bk, kb, Mtok, Dc, Dc, kscale);
  gemm_bf16_kernel<1><<<dim3(Dc * VFc / 128, Mtok / 128), blk, 0, stream>>>(
      xb, Wvb, bv, vtb, Mtok, Dc * VFc, Dc, 1.f);
  gemm_bf16_kernel<0><<<dim3(Dc * VFc / 128, Mtok / 128), blk, 0, stream>>>(
      xb, Wgb, bg, gb, Mtok, Dc * VFc, Dc, 1.f);

  int pairs = Mtok * Dc / 2;
  rotary_kernel<<<(pairs + 255) / 256, 256, 0, stream>>>(qb, sinp, cosp, qrb);
  rotary_kernel<<<(pairs + 255) / 256, 256, 0, stream>>>(kb, sinp, cosp, krb);

  retention_attn_kernel<<<dim3(Lc / 64, Hc, Bc), blk, 0, stream>>>(
      qrb, krb, vtb, gb, lng, lnb, actb);

  gemm_bf16_kernel<2><<<dim3(Dc / 128, Mtok / 128), blk, 0, stream>>>(
      actb, Wob, bo, out, Mtok, Dc, Dc * VFc, 1.f);
}
